// MultiHeadAttention_24824910971155
// MI455X (gfx1250) — matrix operand order repaired and checked by reading, not compiled, not run on hardware
//
#include <hip/hip_runtime.h>
#include <stdint.h>

// ---------------------------------------------------------------------------
// MHA block for MI455X (gfx1250, wave32, WMMA f32_16x16x32_f16, TDM async)
// B=8, L=1024, D=1024, H=16, DH=64, TEMP=8
// ---------------------------------------------------------------------------

typedef __attribute__((ext_vector_type(16))) _Float16     v16h;
typedef __attribute__((ext_vector_type(8)))  _Float16     v8h;
typedef __attribute__((ext_vector_type(8)))  float        v8f;
typedef __attribute__((ext_vector_type(4)))  unsigned int v4u;
typedef __attribute__((ext_vector_type(8)))  int          v8i;
typedef __attribute__((ext_vector_type(4)))  int          v4i;

union AFrag { v16h v; v8h h[2]; };

__device__ __forceinline__ v8f wmma_f16(v16h a, v16h b, v8f c) {
  return __builtin_amdgcn_wmma_f32_16x16x32_f16(
      /*neg_a=*/false, a, /*neg_b=*/false, b,
      /*c_mod=*/(short)0, c, /*reuse_a=*/false, /*reuse_b=*/false);
}

#define MB 8
#define LSEQ 1024
#define DMODEL 1024
#define NHEAD 16
#define DH 64
#define MROWS (MB * LSEQ)          // 8192
#define INV_TEMP 0.125f

// GEMM block tile: 256 M x 64 N, K-step 64. 8 waves; wave owns 32 rows
// (2 m-subtiles), 4 n-subtiles -> 8 accumulator chains, 16 WMMA per K-step.
#define PTM 256
#define PTN 64
#define PTK 64

// ---------------------------------------------------------------------------
// Kernel 1: projection GEMM  out = X @ W   (f32 in, f16 out, head-major store)
// layout 0 -> out[b][h][l][d]   (Q, K)
// layout 1 -> out[b][h][d][l]   (V transposed for contiguous B-frags in P*V)
// ---------------------------------------------------------------------------
__global__ __launch_bounds__(256)
void proj_gemm_kernel(const float* __restrict__ X, const float* __restrict__ W,
                      _Float16* __restrict__ out, int layout) {
  __shared__ _Float16 As[PTM][PTK];   // 32 KB
  __shared__ _Float16 Bs[PTN][PTK];   // 8 KB, W tile transposed [n][k]
  const int tid  = threadIdx.x;
  const int wave = tid >> 5;
  const int lane = tid & 31;
  const int hi   = lane >> 4;
  const int lm   = lane & 15;
  const int row0 = blockIdx.x * PTM;
  const int col0 = blockIdx.y * PTN;

  v8f acc[2][4] = {};

  for (int k0 = 0; k0 < DMODEL; k0 += PTK) {
    if (k0 + PTK < DMODEL)
      __builtin_prefetch(&X[(size_t)(row0 + tid) * DMODEL + k0 + PTK], 0, 1);
    // A tile: 256x64 f32 -> f16  (4096 float4 / 256 thr = 16 each)
    for (int i = tid; i < PTM * (PTK / 4); i += 256) {
      const int r  = i >> 4;
      const int c4 = (i & 15) * 4;
      const float4 f = *(const float4*)&X[(size_t)(row0 + r) * DMODEL + k0 + c4];
      As[r][c4 + 0] = (_Float16)f.x; As[r][c4 + 1] = (_Float16)f.y;
      As[r][c4 + 2] = (_Float16)f.z; As[r][c4 + 3] = (_Float16)f.w;
    }
    // W tile: rows k0..k0+63, cols col0..col0+63, store transposed
    for (int i = tid; i < PTK * (PTN / 4); i += 256) {
      const int kr = i >> 4;
      const int c4 = (i & 15) * 4;
      const float4 f = *(const float4*)&W[(size_t)(k0 + kr) * DMODEL + col0 + c4];
      Bs[c4 + 0][kr] = (_Float16)f.x; Bs[c4 + 1][kr] = (_Float16)f.y;
      Bs[c4 + 2][kr] = (_Float16)f.z; Bs[c4 + 3][kr] = (_Float16)f.w;
    }
    __syncthreads();

#pragma unroll
    for (int kk = 0; kk < PTK; kk += 32) {
      AFrag a0, a1;
      a0.h[0] = *(const v8h*)&As[wave * 32 + lm][kk + hi * 16];
      a0.h[1] = *(const v8h*)&As[wave * 32 + lm][kk + 8 + hi * 16];
      a1.h[0] = *(const v8h*)&As[wave * 32 + 16 + lm][kk + hi * 16];
      a1.h[1] = *(const v8h*)&As[wave * 32 + 16 + lm][kk + 8 + hi * 16];
      v16h b[4];
#pragma unroll
      for (int nt = 0; nt < 4; ++nt)
        b[nt] = *(const v16h*)&Bs[nt * 16 + lm][kk + hi * 16];
#pragma unroll
      for (int nt = 0; nt < 4; ++nt) {
        acc[0][nt] = wmma_f16(a0.v, b[nt], acc[0][nt]);
        acc[1][nt] = wmma_f16(a1.v, b[nt], acc[1][nt]);
      }
    }
    __syncthreads();
  }

  // C store with head-major remap
#pragma unroll
  for (int mt = 0; mt < 2; ++mt) {
#pragma unroll
    for (int nt = 0; nt < 4; ++nt) {
#pragma unroll
      for (int i = 0; i < 8; ++i) {
        const int gr = row0 + wave * 32 + mt * 16 + i + 8 * hi;  // 0..8191
        const int gc = col0 + nt * 16 + lm;                      // 0..1023
        const int bb = gr >> 10, ll = gr & 1023;
        const int hh = gc >> 6,  dd = gc & 63;
        const size_t idx = (layout == 0)
            ? ((size_t)((bb * NHEAD + hh) * LSEQ + ll) * DH + dd)
            : ((size_t)((bb * NHEAD + hh) * DH + dd) * LSEQ + ll);
        out[idx] = (_Float16)acc[mt][nt][i];
      }
    }
  }
}

// ---------------------------------------------------------------------------
// Kernel 2: attention.  One block = (b,h, 16 query rows).  128 threads = 4 waves.
// Dynamic LDS: Qs 16x64 f16 (2KB) | Ss 16x1024 f32 (64KB) | Ps 16x1024 f16 (32KB)
// ---------------------------------------------------------------------------
__global__ __launch_bounds__(128)
void attn_kernel(const _Float16* __restrict__ Qh, const _Float16* __restrict__ Kh,
                 const _Float16* __restrict__ VhT,
                 float* __restrict__ attn_out, _Float16* __restrict__ O) {
  extern __shared__ char smem[];
  _Float16* Qs = (_Float16*)smem;                       // [16][64]
  float*    Ss = (float*)(smem + 2048);                 // [16][1024]
  _Float16* Ps = (_Float16*)(smem + 2048 + 65536);      // [16][1024]

  const int bh   = blockIdx.y;              // b*16 + h
  const int row0 = blockIdx.x * 16;
  const int tid  = threadIdx.x;
  const int wave = tid >> 5;                // 0..3
  const int lane = tid & 31;
  const int hi   = lane >> 4;
  const int lm   = lane & 15;

  const _Float16* Q  = Qh  + (size_t)bh * LSEQ * DH;
  const _Float16* K  = Kh  + (size_t)bh * LSEQ * DH;
  const _Float16* Vt = VhT + (size_t)bh * DH * LSEQ;

  // load 16x64 Q tile (unscaled; 1/TEMP folded into softmax)
  {
    const int r = tid >> 3, c = (tid & 7) * 8;
    *(v8h*)&Qs[r * DH + c] = *(const v8h*)&Q[(size_t)(row0 + r) * DH + c];
  }
  __syncthreads();

  // ---- S = Q K^T : 16 key-tiles of 64; wave owns key-subtile of 16 ----
  for (int kt = 0; kt < 16; ++kt) {
    v8f acc = {};
#pragma unroll
    for (int kk = 0; kk < DH; kk += 32) {
      AFrag a;
      a.h[0] = *(const v8h*)&Qs[lm * DH + kk + hi * 16];
      a.h[1] = *(const v8h*)&Qs[lm * DH + kk + 8 + hi * 16];
      const v16h b =
          *(const v16h*)&K[(size_t)(kt * 64 + wave * 16 + lm) * DH + kk + hi * 16];
      acc = wmma_f16(a.v, b, acc);
    }
    const int col = kt * 64 + wave * 16 + lm;
#pragma unroll
    for (int i = 0; i < 8; ++i) Ss[(i + 8 * hi) * LSEQ + col] = acc[i];
  }
  __syncthreads();

  // ---- softmax: wave handles 4 rows, lane-parallel over 1024 cols ----
  for (int rr = 0; rr < 4; ++rr) {
    const int r = wave * 4 + rr;
    float m = -3.0e38f;
    for (int c = lane; c < LSEQ; c += 32) m = fmaxf(m, Ss[r * LSEQ + c] * INV_TEMP);
#pragma unroll
    for (int off = 16; off > 0; off >>= 1) m = fmaxf(m, __shfl_xor(m, off, 32));
    float s = 0.0f;
    for (int c = lane; c < LSEQ; c += 32) s += __expf(Ss[r * LSEQ + c] * INV_TEMP - m);
#pragma unroll
    for (int off = 16; off > 0; off >>= 1) s += __shfl_xor(s, off, 32);
    const float inv = 1.0f / s;
    float* arow = attn_out + ((size_t)bh * LSEQ + row0 + r) * LSEQ;
    for (int c = lane; c < LSEQ; c += 32) {
      const float p = __expf(Ss[r * LSEQ + c] * INV_TEMP - m) * inv;
      arow[c] = p;
      Ps[r * LSEQ + c] = (_Float16)p;
    }
  }
  __syncthreads();

  // ---- O = P V + Q : wave owns d-subtile of 16; C init = qh residual ----
  v8f acc;
#pragma unroll
  for (int i = 0; i < 8; ++i)
    acc[i] = (float)Qs[(i + 8 * hi) * DH + wave * 16 + lm];
  for (int kk = 0; kk < LSEQ; kk += 32) {
    AFrag a;
    a.h[0] = *(const v8h*)&Ps[lm * LSEQ + kk + hi * 16];
    a.h[1] = *(const v8h*)&Ps[lm * LSEQ + kk + 8 + hi * 16];
    const v16h b =
        *(const v16h*)&Vt[(size_t)(wave * 16 + lm) * LSEQ + kk + hi * 16];
    acc = wmma_f16(a.v, b, acc);
  }
  const int bb = bh >> 4, hh = bh & 15;
#pragma unroll
  for (int i = 0; i < 8; ++i) {
    const int ll = row0 + i + 8 * hi;
    O[(size_t)(bb * LSEQ + ll) * DMODEL + hh * DH + wave * 16 + lm] =
        (_Float16)acc[i];
  }
}

// ---------------------------------------------------------------------------
// Kernel 3: FC GEMM  out = relu(O @ Wfc) + q   (f16 A via TDM, f32 W, f32 out)
// A tile (256 rows x 64 f16, row stride 1024 f16) fetched by the Tensor Data
// Mover directly into LDS while the other waves convert the W tile.
// ---------------------------------------------------------------------------
__global__ __launch_bounds__(256)
void fc_gemm_kernel(const _Float16* __restrict__ A, const float* __restrict__ W,
                    const float* __restrict__ resid, float* __restrict__ out) {
  __shared__ _Float16 As[PTM][PTK];   // 32 KB, packed rows (TDM destination)
  __shared__ _Float16 Bs[PTN][PTK];   // 8 KB
  const int tid  = threadIdx.x;
  const int wave = tid >> 5;
  const int lane = tid & 31;
  const int hi   = lane >> 4;
  const int lm   = lane & 15;
  const int row0 = blockIdx.x * PTM;
  const int col0 = blockIdx.y * PTN;

  const uint32_t lds_base = (uint32_t)(uintptr_t)(void*)&As[0][0];

  v8f acc[2][4] = {};

  for (int k0 = 0; k0 < DMODEL; k0 += PTK) {
    // TDM: DMA the 256x64 f16 A tile into LDS (issued once, by wave 0)
    if (wave == 0) {
      const uint64_t ga =
          (uint64_t)(uintptr_t)&A[(size_t)row0 * DMODEL + k0];
      v4u g0;
      g0[0] = 1u;                                   // count=1, normal mode
      g0[1] = lds_base;                             // lds_addr (bytes)
      g0[2] = (uint32_t)ga;                         // global_addr[31:0]
      g0[3] = (uint32_t)((ga >> 32) & 0x1FFFFFFu)   // global_addr[56:32]
              | (2u << 30);                         // type=2 ("image")
      v8i g1;
      g1[0] = 1 << 16;                // data_size=1 (2 bytes/elem)
      g1[1] = (int)(1024u << 16);     // tensor_dim0 = 1024 (bits 79:48, low)
      g1[2] = (int)(8192u << 16);     // dim0 hi=0 | tensor_dim1 = 8192 (low16)
      g1[3] = (int)(64u << 16);       // dim1 hi=0 | tile_dim0 = 64
      g1[4] = 256;                    // tile_dim1 = 256, tile_dim2 = 0
      g1[5] = 1024;                   // tensor_dim0_stride = 1024 (low 32)
      g1[6] = 0;                      // stride0 hi | dim1_stride low
      g1[7] = 0;
      const v4i z4 = {0, 0, 0, 0};
      const v8i z8 = {0, 0, 0, 0, 0, 0, 0, 0};
      __builtin_amdgcn_tensor_load_to_lds(g0, g1, z4, z4, z8, 0);
    }
    // W tile: f32 -> f16, transposed (overlaps with the TDM transfer)
    for (int i = tid; i < PTK * (PTN / 4); i += 256) {
      const int kr = i >> 4;
      const int c4 = (i & 15) * 4;
      const float4 f = *(const float4*)&W[(size_t)(k0 + kr) * DMODEL + col0 + c4];
      Bs[c4 + 0][kr] = (_Float16)f.x; Bs[c4 + 1][kr] = (_Float16)f.y;
      Bs[c4 + 2][kr] = (_Float16)f.z; Bs[c4 + 3][kr] = (_Float16)f.w;
    }
    if (wave == 0) __builtin_amdgcn_s_wait_tensorcnt((short)0);
    __syncthreads();

#pragma unroll
    for (int kk = 0; kk < PTK; kk += 32) {
      AFrag a0, a1;
      a0.h[0] = *(const v8h*)&As[wave * 32 + lm][kk + hi * 16];
      a0.h[1] = *(const v8h*)&As[wave * 32 + lm][kk + 8 + hi * 16];
      a1.h[0] = *(const v8h*)&As[wave * 32 + 16 + lm][kk + hi * 16];
      a1.h[1] = *(const v8h*)&As[wave * 32 + 16 + lm][kk + 8 + hi * 16];
      v16h b[4];
#pragma unroll
      for (int nt = 0; nt < 4; ++nt)
        b[nt] = *(const v16h*)&Bs[nt * 16 + lm][kk + hi * 16];
#pragma unroll
      for (int nt = 0; nt < 4; ++nt) {
        acc[0][nt] = wmma_f16(a0.v, b[nt], acc[0][nt]);
        acc[1][nt] = wmma_f16(a1.v, b[nt], acc[1][nt]);
      }
    }
    __syncthreads();
  }

#pragma unroll
  for (int mt = 0; mt < 2; ++mt) {
#pragma unroll
    for (int nt = 0; nt < 4; ++nt) {
#pragma unroll
      for (int i = 0; i < 8; ++i) {
        const int gr = row0 + wave * 32 + mt * 16 + i + 8 * hi;
        const int gc = col0 + nt * 16 + lm;
        const size_t idx = (size_t)gr * DMODEL + gc;
        out[idx] = fmaxf(acc[mt][nt][i], 0.0f) + resid[idx];
      }
    }
  }
}

// ---------------------------------------------------------------------------
// Kernel 4: instance-norm over last dim (1024) per row
// ---------------------------------------------------------------------------
__global__ __launch_bounds__(256)
void norm_kernel(const float* __restrict__ X, float* __restrict__ out) {
  __shared__ float ps[8], pss[8];
  const int row = blockIdx.x;
  const float* x = X + (size_t)row * DMODEL;
  const int wave = threadIdx.x >> 5, lane = threadIdx.x & 31;
  float s = 0.0f, ss = 0.0f;
  for (int c = threadIdx.x; c < DMODEL; c += 256) {
    const float v = x[c];
    s += v; ss += v * v;
  }
#pragma unroll
  for (int off = 16; off > 0; off >>= 1) {
    s  += __shfl_xor(s, off, 32);
    ss += __shfl_xor(ss, off, 32);
  }
  if (lane == 0) { ps[wave] = s; pss[wave] = ss; }
  __syncthreads();
  float tot = 0.0f, tots = 0.0f;
#pragma unroll
  for (int i = 0; i < 8; ++i) { tot += ps[i]; tots += pss[i]; }
  const float mu  = tot * (1.0f / DMODEL);
  const float var = tots * (1.0f / DMODEL) - mu * mu;
  const float rr  = rsqrtf(var + 1e-6f);
  for (int c = threadIdx.x; c < DMODEL; c += 256)
    out[(size_t)row * DMODEL + c] = (x[c] - mu) * rr;
}

// ---------------------------------------------------------------------------
extern "C" void kernel_launch(void* const* d_in, const int* in_sizes, int n_in,
                              void* d_out, int out_size, void* d_ws, size_t ws_size,
                              hipStream_t stream) {
  const float* q   = (const float*)d_in[0];
  const float* k   = (const float*)d_in[1];
  const float* v   = (const float*)d_in[2];
  const float* Wq  = (const float*)d_in[3];
  const float* Wk  = (const float*)d_in[4];
  const float* Wv  = (const float*)d_in[5];
  const float* Wfc = (const float*)d_in[6];

  float* o_out    = (float*)d_out;                      // [8,1024,1024]
  float* attn_out = o_out + (size_t)MROWS * DMODEL;     // [8,16,1024,1024]

  const size_t HEAD_ELEMS = (size_t)MB * NHEAD * LSEQ * DH;  // 8388608
  _Float16* Qh   = (_Float16*)d_ws;
  _Float16* Kh   = Qh  + HEAD_ELEMS;
  _Float16* VhT  = Kh  + HEAD_ELEMS;
  _Float16* O    = VhT + HEAD_ELEMS;
  float*    Opre = (float*)(O + HEAD_ELEMS);

  const dim3 gemm_grid(MROWS / PTM, DMODEL / PTN);   // 32 x 16

  proj_gemm_kernel<<<gemm_grid, 256, 0, stream>>>(q, Wq, Qh, 0);
  proj_gemm_kernel<<<gemm_grid, 256, 0, stream>>>(k, Wk, Kh, 0);
  proj_gemm_kernel<<<gemm_grid, 256, 0, stream>>>(v, Wv, VhT, 1);

  const size_t attn_lds = 2048 + 65536 + 32768;      // 100352 B < 320KB/WGP
  attn_kernel<<<dim3(LSEQ / 16, MB * NHEAD), 128, attn_lds, stream>>>(
      Qh, Kh, VhT, attn_out, O);

  fc_gemm_kernel<<<gemm_grid, 256, 0, stream>>>(O, Wfc, q, Opre);
  norm_kernel<<<MROWS, 256, 0, stream>>>(Opre, o_out);
}
